// TransformerXLDecoderLayer_37563783970993
// MI455X (gfx1250) — matrix operand order repaired and checked by reading, not compiled, not run on hardware
//
#include <hip/hip_runtime.h>

typedef __attribute__((ext_vector_type(16))) _Float16 h16;
typedef __attribute__((ext_vector_type(8)))  _Float16 h8;
typedef __attribute__((ext_vector_type(4)))  _Float16 h4;
typedef __attribute__((ext_vector_type(8)))  float    f32x8;
typedef __attribute__((ext_vector_type(4)))  float    f32x4;
typedef __attribute__((ext_vector_type(4)))  unsigned int u32x4;
typedef __attribute__((ext_vector_type(8)))  int      i32x8;
typedef __attribute__((ext_vector_type(4)))  int      i32x4;

#define TT 1024
#define BB 4
#define DD 1024
#define MM 1024
#define SS 2048
#define NN 16
#define HDD 64
#define FFD 4096
constexpr float SCALE = 0.03125f;   // 1/sqrt(D), D=1024
constexpr float EPS   = 1e-5f;

#if defined(__has_builtin)
#if __has_builtin(__builtin_amdgcn_tensor_load_to_lds) && \
    __has_builtin(__builtin_amdgcn_s_wait_tensorcnt)
#define HAVE_TDM 1
#endif
#endif

// ---------------------------------------------------------------- WMMA helpers
__device__ __forceinline__ f32x8 wmma_f16(h16 a, h16 b, f32x8 c) {
  return __builtin_amdgcn_wmma_f32_16x16x32_f16(false, a, false, b, (short)0, c,
                                                false, false);
}
__device__ __forceinline__ h16 mk16(h8 a, h8 b) {
  h16 f;
#pragma unroll
  for (int e = 0; e < 8; ++e) { f[e] = a[e]; f[e + 8] = b[e]; }
  return f;
}

// ------------------------------------------------ TDM tile load (128x32 f16)
// Loads a 128-row x 32-half f16 tile (row stride = strideHalves) into LDS,
// padding each 64B row with 32B so rows land at stride 48 halves (bank-safe).
// D# packing per CDNA5 ISA 8.3/8.4: group0 = {count|flags, lds_addr,
// global_addr lo, global_addr hi | type=2}; group1 = {data_size=2B, pad 16DW
// interval / 8DW amount, tensor_dim0=stride, tensor_dim1=2^20, tile=32x128,
// dim0_stride=stride}.
#ifdef HAVE_TDM
__device__ __forceinline__ void tdm_load_tile(const _Float16* g,
                                              unsigned lds_off,
                                              int strideHalves) {
  unsigned long long ga = (unsigned long long)(uintptr_t)g;
  u32x4 g0;
  g0[0] = 1u;                                            // count=1, user mode
  g0[1] = lds_off;                                       // lds_addr (bytes)
  g0[2] = (unsigned)ga;                                  // global_addr[31:0]
  g0[3] = (unsigned)((ga >> 32) & 0x01FFFFFFu) | (2u << 30); // addr[56:32],type=2
  i32x8 g1;
  g1[0] = (1 << 16)      // data_size = 2 bytes
        | (1 << 20)      // pad_enable
        | (3 << 22)      // pad_interval: every 16 DWORDs (64B)
        | (7 << 25);     // pad_amount: 8 DWORDs (32B)
  g1[1] = (int)(((unsigned)strideHalves & 0xffffu) << 16);   // tensor_dim0 lo16
  g1[2] = (int)(((unsigned)strideHalves >> 16) & 0xffffu);   // tensor_dim0 hi16
  g1[3] = (int)(16u | (32u << 16));   // tensor_dim1 = 1<<20, tile_dim0 = 32
  g1[4] = 128;                        // tile_dim1 = 128, tile_dim2 = 0
  g1[5] = strideHalves;               // tensor_dim0_stride lo32
  g1[6] = 0;
  g1[7] = 0;
  i32x4 z4 = {};
#if defined(__clang_major__) && __clang_major__ >= 23
  i32x8 z8 = {};
  __builtin_amdgcn_tensor_load_to_lds(g0, g1, z4, z4, z8, 0);
#else
  __builtin_amdgcn_tensor_load_to_lds(g0, g1, z4, z4, 0);
#endif
}
#endif

__device__ __forceinline__ void copy_tile_fallback(_Float16 (*dst)[48],
                                                   const _Float16* src, int K,
                                                   int k0, int tid) {
  const int r = tid >> 1, seg = tid & 1;
  *(h16*)&dst[r][seg * 16] = *(const h16*)(src + (size_t)r * K + k0 + seg * 16);
}

// ---------------------------------------------------------------- GEMM kernel
// All operands f16: A row-major [Mrows][K]; BT col-major (row = output col).
// MODE 0: qkv   (A = cat,  BT = WqkvT) -> qu,qv,k,vT (f16)
// MODE 1: pos   (A = pe,   BT = WposT) -> pos         (f16)
// MODE 2: outpj (A = ctx,  BT = WoutT) -> pre = x+Av  (f32)
// MODE 3: ff1   (A = h,    BT = W1T)   -> relu(+b1)   (f16)
// MODE 4: ff2   (A = ff,   BT = W2T)   -> pre = h+Av+b2 (f32)
template <int MODE>
__global__ __launch_bounds__(256) void gemm_wmma(
    const _Float16* __restrict__ Ah, const _Float16* __restrict__ BT, int K,
    const float* __restrict__ ubias, const float* __restrict__ vbias,
    const float* __restrict__ addsrc, const float* __restrict__ bias,
    float* __restrict__ outf,
    _Float16* __restrict__ oh0, _Float16* __restrict__ oh1,
    _Float16* __restrict__ oh2, _Float16* __restrict__ oh3)
{
  __shared__ _Float16 As[2][128][48];   // double-buffered A tile (f16)
  __shared__ _Float16 Bs[2][128][48];   // double-buffered B tile (col-major)

  const int tid  = threadIdx.x;
  const int lane = tid & 31, wid = tid >> 5;
  const int wr = wid >> 1, wc = wid & 1;        // 4x2 wave grid
  const int lr = lane & 15, lh = lane >> 4;
  const int row0 = blockIdx.y * 128;
  const int col0 = blockIdx.x * 128;

  const _Float16* Abase = Ah + (size_t)row0 * K;
  const _Float16* Bbase = BT + (size_t)col0 * K;

  f32x8 acc[2][4] = {};
  const int nk = K / 32;

  // prologue: tile 0
#ifdef HAVE_TDM
  if (tid == 0) {
    tdm_load_tile(Abase, (unsigned)(uintptr_t)&As[0][0][0], K);
    tdm_load_tile(Bbase, (unsigned)(uintptr_t)&Bs[0][0][0], K);
  }
  __builtin_amdgcn_s_wait_tensorcnt(0);
#else
  copy_tile_fallback(As[0], Abase, K, 0, tid);
  copy_tile_fallback(Bs[0], Bbase, K, 0, tid);
#endif
  __syncthreads();

  for (int kb = 0; kb < nk; ++kb) {
    const int cur = kb & 1;
    if (kb + 1 < nk) {   // stage next tile while computing this one
#ifdef HAVE_TDM
      if (tid == 0) {
        tdm_load_tile(Abase + (kb + 1) * 32,
                      (unsigned)(uintptr_t)&As[cur ^ 1][0][0], K);
        tdm_load_tile(Bbase + (kb + 1) * 32,
                      (unsigned)(uintptr_t)&Bs[cur ^ 1][0][0], K);
      }
#else
      copy_tile_fallback(As[cur ^ 1], Abase, K, (kb + 1) * 32, tid);
      copy_tile_fallback(Bs[cur ^ 1], Bbase, K, (kb + 1) * 32, tid);
#endif
    }

    h16 af[2];
#pragma unroll
    for (int s = 0; s < 2; ++s) {
      const _Float16* p = &As[cur][wr * 32 + s * 16 + lr][0];
      af[s] = mk16(*(const h8*)(p + lh * 16), *(const h8*)(p + lh * 16 + 8));
    }
    h16 bf[4];
#pragma unroll
    for (int j = 0; j < 4; ++j) {
      const _Float16* p = &Bs[cur][wc * 64 + j * 16 + lr][lh * 16];
      bf[j] = mk16(*(const h8*)p, *(const h8*)(p + 8));
    }
#pragma unroll
    for (int s = 0; s < 2; ++s)
#pragma unroll
      for (int j = 0; j < 4; ++j) acc[s][j] = wmma_f16(af[s], bf[j], acc[s][j]);

#ifdef HAVE_TDM
    __builtin_amdgcn_s_wait_tensorcnt(0);
#endif
    __syncthreads();
  }

  // ---- epilogue ----
#pragma unroll
  for (int s = 0; s < 2; ++s)
#pragma unroll
    for (int j = 0; j < 4; ++j)
#pragma unroll
      for (int r = 0; r < 8; ++r) {
        const int grow = row0 + wr * 32 + s * 16 + r + 8 * lh;
        const int gcol = col0 + wc * 64 + j * 16 + lr;
        const float v = acc[s][j][r];
        if constexpr (MODE == 0) {
          const int srow = grow >> 2, b = grow & 3;
          const int part = gcol >> 10, cc = gcol & 1023;
          const int n = cc >> 6, hd = cc & 63;
          if (part == 0) {
            if (srow >= MM) {
              const size_t o = (((size_t)(b * NN + n)) * TT + (srow - MM)) * HDD + hd;
              oh0[o] = (_Float16)(v + ubias[n * HDD + hd]);   // q + u_bias
              oh1[o] = (_Float16)(v + vbias[n * HDD + hd]);   // q + v_bias
            }
          } else if (part == 1) {
            oh2[(((size_t)(b * NN + n)) * SS + srow) * HDD + hd] = (_Float16)v; // k
          } else {
            oh3[(((size_t)(b * NN + n)) * HDD + hd) * SS + srow] = (_Float16)v; // vT
          }
        } else if constexpr (MODE == 1) {
          const int n = gcol >> 6, hd = gcol & 63;
          oh0[((size_t)n * SS + grow) * HDD + hd] = (_Float16)v;
        } else if constexpr (MODE == 2) {
          outf[(size_t)grow * DD + gcol] = addsrc[(size_t)grow * DD + gcol] + v;
        } else if constexpr (MODE == 3) {
          const float y = v + bias[gcol];
          oh0[(size_t)grow * FFD + gcol] = (_Float16)(y > 0.f ? y : 0.f);
        } else {
          outf[(size_t)grow * DD + gcol] =
              addsrc[(size_t)grow * DD + gcol] + v + bias[gcol];
        }
      }
}

// ------------------------------------------------------------ pre-passes
// elementwise f32 -> f16 with two-source gather ([a ; b] split at `split`)
__global__ __launch_bounds__(256) void to_f16(const float* __restrict__ a,
                                              const float* __restrict__ b,
                                              long long split, long long n,
                                              _Float16* __restrict__ dst) {
  const long long i = ((long long)blockIdx.x * 256 + threadIdx.x) * 4;
  if (i >= n) return;
  const float* s = (i < split) ? (a + i) : (b + (i - split));
  f32x4 v = *(const f32x4*)s;
  h4 o;
#pragma unroll
  for (int e = 0; e < 4; ++e) o[e] = (_Float16)v[e];
  *(h4*)(dst + i) = o;
}

// f32 [R][C] -> f16 [C][R] (weights to col-major for TDM-friendly B tiles)
__global__ __launch_bounds__(256) void transpose_f16(const float* __restrict__ in,
                                                     _Float16* __restrict__ out,
                                                     int R, int C) {
  __shared__ float t[32][33];
  const int bx = blockIdx.x * 32;   // col base
  const int by = blockIdx.y * 32;   // row base
  const int tx = threadIdx.x & 31, ty = threadIdx.x >> 5;
#pragma unroll
  for (int q = 0; q < 4; ++q)
    t[ty + q * 8][tx] = in[(size_t)(by + ty + q * 8) * C + bx + tx];
  __syncthreads();
#pragma unroll
  for (int q = 0; q < 4; ++q)
    out[(size_t)(bx + ty + q * 8) * R + by + tx] = (_Float16)t[tx][ty + q * 8];
}

// ------------------------------------------------------- fused attention
// One block = 16 query rows x one (b,n). Full 16xS score row lives in LDS
// (128 KB -- possible thanks to CDNA5's 320 KB WGP LDS).
__global__ __launch_bounds__(128) void attn_fused(
    const _Float16* __restrict__ qu, const _Float16* __restrict__ qv,
    const _Float16* __restrict__ kf, const _Float16* __restrict__ vT,
    const _Float16* __restrict__ posf,
    float* __restrict__ attnO, _Float16* __restrict__ ctx)
{
  extern __shared__ float smem[];
  float* Sbuf = smem;                    // 16 * 2048 scores/probs
  float* Pbuf = Sbuf + 16 * SS;          // 4 waves * 16 * 96 pos window / ctx partials
  float* Red  = Pbuf + 4 * 16 * 96;      // 16 * 8 reduction scratch

  const int i0 = blockIdx.x * 16;
  const int bn = blockIdx.y;
  const int b = bn >> 4, n = bn & 15;
  const int tid = threadIdx.x;
  const int lane = tid & 31, w = tid >> 5;
  const int lr = lane & 15, lh = lane >> 4;

  const _Float16* quB = qu + (((size_t)(b * NN + n)) * TT + i0) * HDD;
  const _Float16* qvB = qv + (((size_t)(b * NN + n)) * TT + i0) * HDD;
  const _Float16* kB  = kf + ((size_t)(b * NN + n)) * SS * HDD;
  const _Float16* vB  = vT + ((size_t)(b * NN + n)) * HDD * SS;
  const _Float16* pB  = posf + (size_t)n * SS * HDD;

  h16 aU[2], aV[2];
#pragma unroll
  for (int ks = 0; ks < 2; ++ks) {
    const _Float16* p = quB + lr * HDD + ks * 32;
    aU[ks] = mk16(*(const h8*)(p + lh * 16), *(const h8*)(p + lh * 16 + 8));
    const _Float16* q = qvB + lr * HDD + ks * 32;
    aV[ks] = mk16(*(const h8*)(q + lh * 16), *(const h8*)(q + lh * 16 + 8));
  }

  float* Pw = Pbuf + w * 16 * 96;
  for (int ch = w; ch < SS / 64; ch += 4) {       // wave-private 64-col chunks
    const int j0 = ch * 64;
#pragma unroll
    for (int jt = 0; jt < 4; ++jt) {              // content = (q+u) . k
      f32x8 a = {};
#pragma unroll
      for (int ks = 0; ks < 2; ++ks) {
        h16 bK = *(const h16*)(kB + (size_t)(j0 + jt * 16 + lr) * HDD + ks * 32 + lh * 16);
        a = wmma_f16(aU[ks], bK, a);
      }
      const int colb = j0 + jt * 16 + lr;
#pragma unroll
      for (int r = 0; r < 8; ++r) Sbuf[(size_t)(r + 8 * lh) * SS + colb] = a[r];
    }
    const int w0 = j0 + (TT - 1) - i0 - 15;       // shifted pos window [w0, w0+95]
#pragma unroll
    for (int pt = 0; pt < 6; ++pt) {
      const int jj = w0 + pt * 16 + lr;
      f32x8 a = {};
#pragma unroll
      for (int ks = 0; ks < 2; ++ks) {
        h16 bP = {};
        if (jj < SS) bP = *(const h16*)(pB + (size_t)jj * HDD + ks * 32 + lh * 16);
        a = wmma_f16(aV[ks], bP, a);
      }
#pragma unroll
      for (int r = 0; r < 8; ++r) Pw[(r + 8 * lh) * 96 + pt * 16 + lr] = a[r];
    }
#pragma unroll
    for (int it = 0; it < 32; ++it) {             // combine + rel-shift + mask
      const int e = it * 32 + lane;
      const int ii = e >> 6, c = e & 63;
      const int j = j0 + c;
      float sv = (Sbuf[(size_t)ii * SS + j] + Pw[ii * 96 + c + 15 - ii]) * SCALE;
      if (j > i0 + ii + MM) sv = -INFINITY;
      Sbuf[(size_t)ii * SS + j] = sv;
    }
  }
  __syncthreads();

  { // exact softmax over full row, write normalized attn
    const int ii = tid >> 3, t8 = tid & 7;
    float* row = Sbuf + (size_t)ii * SS;
    float mx = -INFINITY;
    for (int j = t8; j < SS; j += 8) mx = fmaxf(mx, row[j]);
    Red[ii * 8 + t8] = mx;
    __syncthreads();
    mx = Red[ii * 8];
#pragma unroll
    for (int q = 1; q < 8; ++q) mx = fmaxf(mx, Red[ii * 8 + q]);
    __syncthreads();
    float sm = 0.f;
    for (int j = t8; j < SS; j += 8) {
      const float pv = __expf(row[j] - mx);
      row[j] = pv; sm += pv;
    }
    Red[ii * 8 + t8] = sm;
    __syncthreads();
    sm = 0.f;
#pragma unroll
    for (int q = 0; q < 8; ++q) sm += Red[ii * 8 + q];
    const float inv = 1.0f / sm;
    const int t = i0 + ii;
    for (int j = t8; j < SS; j += 8) {
      const float pv = row[j] * inv;
      row[j] = pv;
      attnO[(((size_t)t * SS + j) * BB + b) * NN + n] = pv;
    }
  }
  __syncthreads();

  // context = attn @ v  (K = S split across 4 waves)
  f32x8 cacc[4] = {};
  for (int k0 = w * 32; k0 < SS; k0 += 128) {
    h16 aP;
    const float* pr = Sbuf + (size_t)lr * SS + k0;
#pragma unroll
    for (int e = 0; e < 8; ++e) {
      aP[e]     = (_Float16)pr[lh * 16 + e];
      aP[e + 8] = (_Float16)pr[lh * 16 + 8 + e];
    }
#pragma unroll
    for (int ht = 0; ht < 4; ++ht) {
      h16 bV = *(const h16*)(vB + (size_t)(ht * 16 + lr) * SS + k0 + lh * 16);
      cacc[ht] = wmma_f16(aP, bV, cacc[ht]);
    }
  }
  float* Cw = Pbuf + w * 16 * 64;
#pragma unroll
  for (int ht = 0; ht < 4; ++ht)
#pragma unroll
    for (int r = 0; r < 8; ++r) Cw[(r + 8 * lh) * 64 + ht * 16 + lr] = cacc[ht][r];
  __syncthreads();
  for (int e = tid; e < 16 * 64; e += 128) {
    const int ii = e >> 6, hd = e & 63;
    const float s = Pbuf[ii * 64 + hd] + Pbuf[1 * 1024 + ii * 64 + hd] +
                    Pbuf[2 * 1024 + ii * 64 + hd] + Pbuf[3 * 1024 + ii * 64 + hd];
    ctx[((size_t)(i0 + ii) * BB + b) * (NN * HDD) + n * HDD + hd] = (_Float16)s;
  }
}

// ---------------------------------------------------------------- layer norm
__global__ __launch_bounds__(256) void ln_kernel(
    const float* __restrict__ in, const float* __restrict__ g,
    const float* __restrict__ be, float* __restrict__ outf,
    _Float16* __restrict__ outh)
{
  __shared__ float red[256];
  const int row = blockIdx.x;
  const int t = threadIdx.x;
  const float* p = in + (size_t)row * DD;
  float s = 0.f;
  for (int c = t; c < DD; c += 256) s += p[c];
  red[t] = s; __syncthreads();
  for (int st = 128; st > 0; st >>= 1) {
    if (t < st) red[t] += red[t + st];
    __syncthreads();
  }
  const float mu = red[0] * (1.0f / DD);
  __syncthreads();
  float v = 0.f;
  for (int c = t; c < DD; c += 256) { const float d = p[c] - mu; v += d * d; }
  red[t] = v; __syncthreads();
  for (int st = 128; st > 0; st >>= 1) {
    if (t < st) red[t] += red[t + st];
    __syncthreads();
  }
  const float rstd = rsqrtf(red[0] * (1.0f / DD) + EPS);
  for (int c = t; c < DD; c += 256) {
    const float y = (p[c] - mu) * rstd * g[c] + be[c];
    if (outf) outf[(size_t)row * DD + c] = y;
    if (outh) outh[(size_t)row * DD + c] = (_Float16)y;
  }
}

// ---------------------------------------------------------------- launcher
extern "C" void kernel_launch(void* const* d_in, const int* in_sizes, int n_in,
                              void* d_out, int out_size, void* d_ws, size_t ws_size,
                              hipStream_t stream) {
  (void)in_sizes; (void)n_in; (void)out_size; (void)ws_size;
  const float* x    = (const float*)d_in[0];
  const float* pe   = (const float*)d_in[1];
  const float* ub   = (const float*)d_in[2];
  const float* vb   = (const float*)d_in[3];
  const float* mems = (const float*)d_in[4];
  // d_in[5] attn_mask: causal mask recomputed analytically (j > i + M)
  const float* Wqkv = (const float*)d_in[6];
  const float* Wpos = (const float*)d_in[7];
  const float* Wout = (const float*)d_in[8];
  const float* W1   = (const float*)d_in[9];
  const float* b1   = (const float*)d_in[10];
  const float* W2   = (const float*)d_in[11];
  const float* b2   = (const float*)d_in[12];
  const float* g1   = (const float*)d_in[13];
  const float* be1  = (const float*)d_in[14];
  const float* g2   = (const float*)d_in[15];
  const float* be2  = (const float*)d_in[16];

  char* base = (char*)d_ws;
  size_t off = 0;
  auto carve = [&](size_t bytes) -> char* {
    char* p = base + off;
    off = (off + bytes + 255) & ~(size_t)255;
    return p;
  };
  _Float16* quW  = (_Float16*)carve((size_t)BB * NN * TT * HDD * 2);
  _Float16* qvW  = (_Float16*)carve((size_t)BB * NN * TT * HDD * 2);
  _Float16* kW   = (_Float16*)carve((size_t)BB * NN * SS * HDD * 2);
  _Float16* vTW  = (_Float16*)carve((size_t)BB * NN * HDD * SS * 2);
  _Float16* posW = (_Float16*)carve((size_t)NN * SS * HDD * 2);
  _Float16* ctxW = (_Float16*)carve((size_t)TT * BB * NN * HDD * 2);
  float*    pre  = (float*)   carve((size_t)TT * BB * DD * 4);   // pre-LN (reused)
  float*    hf   = (float*)   carve((size_t)TT * BB * DD * 4);   // h fp32
  _Float16* hh   = (_Float16*)carve((size_t)TT * BB * DD * 2);   // h f16
  _Float16* ffW  = (_Float16*)carve((size_t)TT * BB * FFD * 2);  // ff f16
  _Float16* catW = (_Float16*)carve((size_t)SS * BB * DD * 2);   // [mems;x] f16
  _Float16* peW  = (_Float16*)carve((size_t)SS * DD * 2);        // pos_enc f16
  _Float16* WqkvT= (_Float16*)carve((size_t)3 * NN * HDD * DD * 2);
  _Float16* WposT= (_Float16*)carve((size_t)NN * HDD * DD * 2);
  _Float16* WoutT= (_Float16*)carve((size_t)DD * NN * HDD * 2);
  _Float16* W1T  = (_Float16*)carve((size_t)FFD * DD * 2);
  _Float16* W2T  = (_Float16*)carve((size_t)DD * FFD * 2);

  float* outP  = (float*)d_out;
  float* attnP = outP + (size_t)TT * BB * DD;

  // 0) precision pre-passes: activations -> f16, weights -> f16 col-major
  {
    const long long ncat = (long long)SS * BB * DD;
    to_f16<<<(unsigned)(ncat / 1024), 256, 0, stream>>>(
        mems, x, (long long)MM * BB * DD, ncat, catW);
    const long long npe = (long long)SS * DD;
    to_f16<<<(unsigned)(npe / 1024), 256, 0, stream>>>(pe, pe, npe, npe, peW);
    transpose_f16<<<dim3(3 * NN * HDD / 32, DD / 32), 256, 0, stream>>>(
        Wqkv, WqkvT, DD, 3 * NN * HDD);
    transpose_f16<<<dim3(NN * HDD / 32, DD / 32), 256, 0, stream>>>(
        Wpos, WposT, DD, NN * HDD);
    transpose_f16<<<dim3(DD / 32, NN * HDD / 32), 256, 0, stream>>>(
        Wout, WoutT, NN * HDD, DD);
    transpose_f16<<<dim3(FFD / 32, DD / 32), 256, 0, stream>>>(W1, W1T, DD, FFD);
    transpose_f16<<<dim3(DD / 32, FFD / 32), 256, 0, stream>>>(W2, W2T, FFD, DD);
  }
  // 1) QKV projection
  gemm_wmma<0><<<dim3(3 * NN * HDD / 128, SS * BB / 128), 256, 0, stream>>>(
      catW, WqkvT, DD, ub, vb, nullptr, nullptr,
      nullptr, quW, qvW, kW, vTW);
  // 2) positional projection
  gemm_wmma<1><<<dim3(NN * HDD / 128, SS / 128), 256, 0, stream>>>(
      peW, WposT, DD, nullptr, nullptr, nullptr, nullptr,
      nullptr, posW, nullptr, nullptr, nullptr);
  // 3) fused attention (scores + rel-shift + mask + softmax + attn-out + context)
  const size_t smem = (size_t)(16 * SS + 4 * 16 * 96 + 16 * 8) * 4;
  attn_fused<<<dim3(TT / 16, BB * NN), 128, smem, stream>>>(
      quW, qvW, kW, vTW, posW, attnP, ctxW);
  // 4) output projection + residual
  gemm_wmma<2><<<dim3(DD / 128, TT * BB / 128), 256, 0, stream>>>(
      ctxW, WoutT, NN * HDD, nullptr, nullptr, x, nullptr,
      pre, nullptr, nullptr, nullptr, nullptr);
  // 5) LayerNorm 1
  ln_kernel<<<TT * BB, 256, 0, stream>>>(pre, g1, be1, hf, hh);
  // 6) FFN expand (+b1, ReLU)
  gemm_wmma<3><<<dim3(FFD / 128, TT * BB / 128), 256, 0, stream>>>(
      hh, W1T, DD, nullptr, nullptr, nullptr, b1,
      nullptr, ffW, nullptr, nullptr, nullptr);
  // 7) FFN contract (+b2) + residual
  gemm_wmma<4><<<dim3(DD / 128, TT * BB / 128), 256, 0, stream>>>(
      ffW, W2T, FFD, nullptr, nullptr, hf, b2,
      pre, nullptr, nullptr, nullptr, nullptr);
  // 8) LayerNorm 2 -> out
  ln_kernel<<<TT * BB, 256, 0, stream>>>(pre, g2, be2, outP, nullptr);
}
